// CommAttention_66589172957644
// MI455X (gfx1250) — matrix operand order repaired and checked by reading, not compiled, not run on hardware
//
#include <hip/hip_runtime.h>
#include <hip/hip_bf16.h>

typedef __attribute__((ext_vector_type(16))) _Float16 v16h;
typedef __attribute__((ext_vector_type(8)))  _Float16 v8h;
typedef __attribute__((ext_vector_type(4)))  _Float16 h4;
typedef __attribute__((ext_vector_type(8)))  float    v8f;

union FragU { v8h p2[2]; v16h v; };
struct Acc8 { union { float f[8]; v8f v; }; };

__device__ __forceinline__ v8f vzero8() { v8f z = {}; return z; }

// A fragment: logical A[m][k] row-major p[m*ld + k]; two contiguous 16B runs.
__device__ __forceinline__ v16h load_a(const _Float16* p, int ld, int rowBase, int colBase, int lane) {
    int m = rowBase + (lane & 15);
    int hioff = (lane >= 16) ? 8 : 0;
    const _Float16* base = p + m * ld + colBase + hioff;
    FragU u;
    u.p2[0] = *(const v8h*)(base);
    u.p2[1] = *(const v8h*)(base + 16);
    return u.v;
}

// B fragment, K-contiguous storage: logical B[k][n] = p[n*ld + k]; two contiguous 16B runs.
__device__ __forceinline__ v16h load_ct(const _Float16* p, int ld, int kBase, int nBase, int lane) {
    int n = nBase + (lane & 15);
    int k0 = kBase + ((lane >= 16) ? 8 : 0);
    const _Float16* base = p + n * ld + k0;
    FragU u;
    u.p2[0] = *(const v8h*)(base);
    u.p2[1] = *(const v8h*)(base + 16);
    return u.v;
}

__device__ __forceinline__ v8f wmma32(v16h a, v16h b, v8f c) {
    return __builtin_amdgcn_wmma_f32_16x16x32_f16(false, a, false, b, (short)0, c, false, false);
}

// CDNA5 async global->LDS copy (16B per lane), tracked by ASYNCcnt.
__device__ __forceinline__ void async_copy16(unsigned lds_byte, const _Float16* gsrc) {
    unsigned long long ga = (unsigned long long)(uintptr_t)gsrc;
    asm volatile("global_load_async_to_lds_b128 %0, %1, off"
                 :: "v"(lds_byte), "v"(ga) : "memory");
}
__device__ __forceinline__ void wait_async0() {
    asm volatile("s_wait_asynccnt 0" ::: "memory");
}
__device__ __forceinline__ unsigned lds_addr(const void* p) {
    return (unsigned)(uintptr_t)p;
}

// ---------------------------------------------------------------------------
// Prep kernels: f32 -> f16 conversions and per-head weight transposes.
// ---------------------------------------------------------------------------
__global__ __launch_bounds__(256) void cvt16_kernel(const float* __restrict__ src,
                                                    _Float16* __restrict__ dst, int n4) {
    int i = blockIdx.x * 256 + threadIdx.x;
    if (i < n4) {
        float4 f = ((const float4*)src)[i];
        h4 o = { (_Float16)f.x, (_Float16)f.y, (_Float16)f.z, (_Float16)f.w };
        ((h4*)dst)[i] = o;
    }
}

// dst[h][n][k] = src[k][h*128+n]  (src: 128 x 1024)
__global__ __launch_bounds__(128) void wtrans_kernel(const float* __restrict__ src,
                                                     _Float16* __restrict__ dst) {
    int hn = blockIdx.x;            // 0..1023 : h = hn>>7, n = hn&127
    int k  = threadIdx.x;           // 0..127
    int h = hn >> 7, n = hn & 127;
    dst[(size_t)hn * 128 + k] = (_Float16)src[(size_t)k * 1024 + h * 128 + n];
}

// dst[n][k] = src[k][n]  (src: 1024 x 128)
__global__ __launch_bounds__(256) void wotrans_kernel(const float* __restrict__ src,
                                                      _Float16* __restrict__ dst) {
    int n = blockIdx.x;             // 0..127
    for (int k = threadIdx.x; k < 1024; k += 256)
        dst[(size_t)n * 1024 + k] = (_Float16)src[(size_t)k * 128 + n];
}

// ---------------------------------------------------------------------------
// Kernel A: te = mish(t_embed); tb_* = te @ W*t + b*t   (256 blocks of (b,t))
// ---------------------------------------------------------------------------
__global__ __launch_bounds__(128) void bias_kernel(
    const float* __restrict__ t_embed,
    const float* __restrict__ Wqt, const float* __restrict__ bqt,
    const float* __restrict__ Wkt, const float* __restrict__ bkt,
    const float* __restrict__ Wvt, const float* __restrict__ bvt,
    float* __restrict__ tbq, float* __restrict__ tbk, float* __restrict__ tbv)
{
    constexpr int F_ = 128, HID_ = 1024;
    __shared__ float te[F_];
    int bt = blockIdx.x;
    int i  = threadIdx.x;
    float x  = t_embed[(size_t)bt * F_ + i];
    float sp = (x > 20.f) ? x : log1pf(__expf(x));
    te[i] = x * tanhf(sp);
    __syncthreads();
    const float* Ws[3] = {Wqt, Wkt, Wvt};
    const float* bs[3] = {bqt, bkt, bvt};
    float*       os[3] = {tbq, tbk, tbv};
#pragma unroll
    for (int m = 0; m < 3; ++m) {
        for (int rep = 0; rep < 8; ++rep) {
            int n = rep * F_ + i;
            float acc = bs[m][n];
            for (int k = 0; k < F_; ++k) acc += te[k] * Ws[m][(size_t)k * HID_ + n];
            os[m][(size_t)bt * HID_ + n] = acc;
        }
    }
}

// ---------------------------------------------------------------------------
// Kernel B: fused flash attention per (b, l, h). 512 blocks x 256 threads.
// Wk_h / Wv_h persist in LDS; all staging is async global->LDS b128.
// ---------------------------------------------------------------------------
__global__ __launch_bounds__(256) void attn_kernel(
    const _Float16* __restrict__ qh, const _Float16* __restrict__ kvh,
    const _Float16* __restrict__ WqT, const _Float16* __restrict__ WkT,
    const _Float16* __restrict__ WvT,
    const float* __restrict__ tbq, const float* __restrict__ tbk, const float* __restrict__ tbv,
    _Float16* __restrict__ attn_out)
{
    constexpr int T_ = 64, F_ = 128, HID_ = 1024, L_ = 16, N_ = 16, TK_ = 64, LP_ = 32;
    constexpr int SC = 64;
    constexpr int NCHUNK = (TK_ * LP_) / SC;     // 32
    const float scale = 0.08838834764831845f;    // 1/sqrt(128)

    int bl = blockIdx.x;
    int h = bl & 7; int t2 = bl >> 3; int l = t2 & 15; int b = t2 >> 4;

    int tid  = threadIdx.x;
    int lane = tid & 31;
    int w    = tid >> 5;
    int hi   = lane >> 4;
    int ln   = lane & 15;
    int mt   = w >> 1;
    int half = w & 1;

    __shared__ _Float16 sK [SC * F_];    // 16KB K chunk [s][d] (temp f16 Q first)
    __shared__ _Float16 sVT[F_ * SC];    // 16KB V chunk transposed [d][s]
    __shared__ float    sS [T_ * SC];    // 16KB scores (f16 staging view too)
    __shared__ _Float16 sWT[F_ * 32];    //  8KB Wq slice [n][k] (P view too)
    __shared__ _Float16 sWk[F_ * F_];    // 32KB persistent Wk_h [n][k]
    __shared__ _Float16 sWv[F_ * F_];    // 32KB persistent Wv_h [n][k]
    __shared__ float sScale[T_], sM[T_], sL[T_], sPart[256];

    _Float16* sStage = (_Float16*)sS;
    _Float16* sP     = sWT;
    unsigned ldsStage = lds_addr(sStage);
    unsigned ldsWT    = lds_addr(sWT);
    unsigned ldsWk    = lds_addr(sWk);
    unsigned ldsWv    = lds_addr(sWv);

    const _Float16* wq_h = WqT + (size_t)h * 128 * 128;
    const _Float16* wk_h = WkT + (size_t)h * 128 * 128;
    const _Float16* wv_h = WvT + (size_t)h * 128 * 128;

    // ---- Stage q (64x128) + persistent Wk/Wv slabs (128x128 each), one wait ----
    {
        const _Float16* qp = qh + ((size_t)(b * T_) * L_ + l) * F_;
        for (int i = tid; i < T_ * 16; i += 256) {
            int t = i >> 4, seg = i & 15;
            async_copy16(ldsStage + (unsigned)(t * F_ + seg * 8) * 2,
                         qp + (size_t)t * L_ * F_ + seg * 8);
        }
        for (int i = tid; i < F_ * 16; i += 256) {   // 128 rows x 16 segs
            int n = i >> 4, seg = i & 15;
            unsigned off = (unsigned)(n * F_ + seg * 8) * 2;
            async_copy16(ldsWk + off, wk_h + (size_t)n * F_ + seg * 8);
            async_copy16(ldsWv + off, wv_h + (size_t)n * F_ + seg * 8);
        }
        wait_async0();
    }
    __syncthreads();

    // ---- Q = q @ Wq_h + tb_q, scaled -> sK (f16) ----
    Acc8 qacc[4];
#pragma unroll
    for (int j = 0; j < 4; ++j) qacc[j].v = vzero8();
    for (int kk = 0; kk < 4; ++kk) {
        for (int i = tid; i < 512; i += 256) {           // 128 n x 4 segs
            int n = i >> 2, seg = i & 3;
            async_copy16(ldsWT + (unsigned)(n * 32 + seg * 8) * 2,
                         wq_h + (size_t)n * 128 + kk * 32 + seg * 8);
        }
        wait_async0();
        __syncthreads();
        v16h a = load_a(sStage, F_, mt * 16, kk * 32, lane);
#pragma unroll
        for (int j = 0; j < 4; ++j) {
            v16h bf = load_ct(sWT, 32, 0, (half * 4 + j) * 16, lane);
            qacc[j].v = wmma32(a, bf, qacc[j].v);
        }
        __syncthreads();
    }
    {
        const float* tq = tbq + (size_t)(b * T_) * HID_ + h * F_;
#pragma unroll
        for (int j = 0; j < 4; ++j) {
            int col = (half * 4 + j) * 16 + ln;
#pragma unroll
            for (int r = 0; r < 8; ++r) {
                int row = mt * 16 + hi * 8 + r;
                float v = (qacc[j].f[r] + tq[(size_t)row * HID_ + col]) * scale;
                sK[row * F_ + col] = (_Float16)v;
            }
        }
    }
    if (tid < T_) { sM[tid] = -3.0e38f; sL[tid] = 0.f; }
    __syncthreads();

    v16h aQ[4];
#pragma unroll
    for (int kk = 0; kk < 4; ++kk) aQ[kk] = load_a(sK, F_, mt * 16, kk * 32, lane);

    Acc8 oacc[4];
#pragma unroll
    for (int j = 0; j < 4; ++j) oacc[j].v = vzero8();
    __syncthreads();

    const _Float16* kvp = kvh + (size_t)(b * N_ + l) * TK_ * LP_ * F_;

    for (int c = 0; c < NCHUNK; ++c) {
        // 1. stage kv chunk (64 x 128 f16) async
        for (int i = tid; i < SC * 16; i += 256) {
            int r = i >> 4, seg = i & 15;
            async_copy16(ldsStage + (unsigned)(r * F_ + seg * 8) * 2,
                         kvp + (size_t)(c * SC + r) * F_ + seg * 8);
        }
        if (c + 1 < NCHUNK)
            __builtin_prefetch(kvp + (size_t)(c + 1) * SC * F_ + (size_t)tid * 16, 0, 1);
        wait_async0();
        __syncthreads();

        // preload kv A-fragments for this wave's M tile (sStage freed at next barrier)
        v16h aKV[4];
#pragma unroll
        for (int kk = 0; kk < 4; ++kk) aKV[kk] = load_a(sStage, F_, mt * 16, kk * 32, lane);

        // 2/3. K and V projections from persistent weight slabs (no staging)
        {
            Acc8 kacc[4], vacc[4];
#pragma unroll
            for (int j = 0; j < 4; ++j) { kacc[j].v = vzero8(); vacc[j].v = vzero8(); }
#pragma unroll
            for (int kk = 0; kk < 4; ++kk) {
#pragma unroll
                for (int j = 0; j < 4; ++j) {
                    int nb = (half * 4 + j) * 16;
                    v16h bk = load_ct(sWk, F_, kk * 32, nb, lane);
                    kacc[j].v = wmma32(aKV[kk], bk, kacc[j].v);
                    v16h bv = load_ct(sWv, F_, kk * 32, nb, lane);
                    vacc[j].v = wmma32(aKV[kk], bv, vacc[j].v);
                }
            }
            const float* tbkp = tbk + (size_t)(b * TK_) * HID_ + h * F_;
            const float* tbvp = tbv + (size_t)(b * TK_) * HID_ + h * F_;
#pragma unroll
            for (int j = 0; j < 4; ++j) {
                int col = (half * 4 + j) * 16 + ln;
#pragma unroll
                for (int r = 0; r < 8; ++r) {
                    int rowc = mt * 16 + hi * 8 + r;
                    int tk   = (c * SC + rowc) >> 5;
                    sK [rowc * F_ + col] = (_Float16)(kacc[j].f[r] + tbkp[(size_t)tk * HID_ + col]);
                    sVT[col * SC + rowc] = (_Float16)(vacc[j].f[r] + tbvp[(size_t)tk * HID_ + col]);
                }
            }
        }
        __syncthreads();

        // 4. S = Qs @ K^T (64x64)
        Acc8 sacc[2];
        sacc[0].v = vzero8(); sacc[1].v = vzero8();
#pragma unroll
        for (int kk = 0; kk < 4; ++kk) {
#pragma unroll
            for (int j = 0; j < 2; ++j) {
                v16h bf = load_ct(sK, F_, kk * 32, (half * 2 + j) * 16, lane);
                sacc[j].v = wmma32(aQ[kk], bf, sacc[j].v);
            }
        }
#pragma unroll
        for (int j = 0; j < 2; ++j) {
            int col = (half * 2 + j) * 16 + ln;
#pragma unroll
            for (int r = 0; r < 8; ++r)
                sS[(mt * 16 + hi * 8 + r) * SC + col] = sacc[j].f[r];
        }
        __syncthreads();

        // 5. online softmax (4 threads per row, vectorized LDS access)
        {
            int row = tid >> 2, sub = tid & 3;
            const float4* srow = (const float4*)(sS + row * SC + sub * 16);
            float mx = -3.0e38f;
#pragma unroll
            for (int q4 = 0; q4 < 4; ++q4) {
                float4 fv = srow[q4];
                mx = fmaxf(mx, fmaxf(fmaxf(fv.x, fv.y), fmaxf(fv.z, fv.w)));
            }
            sPart[tid] = mx;
            __syncthreads();
            if (sub == 0) {
                float cm = fmaxf(fmaxf(sPart[tid], sPart[tid + 1]),
                                 fmaxf(sPart[tid + 2], sPart[tid + 3]));
                float mo = sM[row];
                float mn = fmaxf(mo, cm);
                sM[row] = mn;
                float sc = __expf(mo - mn);
                sScale[row] = sc;
                sL[row] *= sc;
            }
            __syncthreads();
            float mn = sM[row];
            float ps = 0.f;
#pragma unroll
            for (int q4 = 0; q4 < 4; ++q4) {
                float4 fv = srow[q4];
                float p0 = __expf(fv.x - mn);
                float p1 = __expf(fv.y - mn);
                float p2 = __expf(fv.z - mn);
                float p3 = __expf(fv.w - mn);
                h4 o = { (_Float16)p0, (_Float16)p1, (_Float16)p2, (_Float16)p3 };
                *(h4*)(sP + row * SC + sub * 16 + q4 * 4) = o;
                ps += (p0 + p1) + (p2 + p3);
            }
            sPart[tid] = ps;
            __syncthreads();
            if (sub == 0)
                sL[row] += sPart[tid] + sPart[tid + 1] + sPart[tid + 2] + sPart[tid + 3];
            __syncthreads();
        }

        // 6. rescale O
#pragma unroll
        for (int j = 0; j < 4; ++j)
#pragma unroll
            for (int r = 0; r < 8; ++r)
                oacc[j].f[r] *= sScale[mt * 16 + hi * 8 + r];

        // 7. O += P @ V
#pragma unroll
        for (int kk = 0; kk < 2; ++kk) {
            v16h aP = load_a(sP, SC, mt * 16, kk * 32, lane);
#pragma unroll
            for (int j = 0; j < 4; ++j) {
                v16h bf = load_ct(sVT, SC, kk * 32, (half * 4 + j) * 16, lane);
                oacc[j].v = wmma32(aP, bf, oacc[j].v);
            }
        }
    }

    __syncthreads();
    // ---- normalize & store head output (f16) ----
#pragma unroll
    for (int j = 0; j < 4; ++j) {
        int col = (half * 4 + j) * 16 + ln;
#pragma unroll
        for (int r = 0; r < 8; ++r) {
            int row = mt * 16 + hi * 8 + r;
            float ov = oacc[j].f[r] / sL[row];
            attn_out[((size_t)(b * T_ + row) * L_ + l) * HID_ + h * F_ + col] = (_Float16)ov;
        }
    }
}

// ---------------------------------------------------------------------------
// Kernel C: out = attn @ WoT^T + bo.  64 blocks x 256 threads, 64 rows each.
// ---------------------------------------------------------------------------
__global__ __launch_bounds__(256) void outproj_kernel(
    const _Float16* __restrict__ attn, const _Float16* __restrict__ WoT,
    const float* __restrict__ bo, float* __restrict__ out)
{
    constexpr int HID_ = 1024, F_ = 128;
    __shared__ _Float16 sA [64 * 32];    // A slice row-major [row][k]
    __shared__ _Float16 sBT[F_ * 32];    // Wo slice [n][k]
    int tid = threadIdx.x, lane = tid & 31, w = tid >> 5;
    int hi = lane >> 4, ln = lane & 15;
    int mt = w >> 1, half = w & 1;
    size_t rbase = (size_t)blockIdx.x * 64;
    unsigned ldsA  = lds_addr(sA);
    unsigned ldsBT = lds_addr(sBT);

    Acc8 acc[4];
#pragma unroll
    for (int j = 0; j < 4; ++j) acc[j].v = vzero8();

    for (int kk = 0; kk < 32; ++kk) {
        {   // A slice: 64 rows x 4 segs (async b128)
            int r = tid >> 2, seg = tid & 3;
            async_copy16(ldsA + (unsigned)(r * 32 + seg * 8) * 2,
                         attn + (rbase + r) * HID_ + kk * 32 + seg * 8);
        }
        for (int i = tid; i < 512; i += 256) {  // Wo slice: 128 n x 4 segs
            int n = i >> 2, seg = i & 3;
            async_copy16(ldsBT + (unsigned)(n * 32 + seg * 8) * 2,
                         WoT + (size_t)n * HID_ + kk * 32 + seg * 8);
        }
        if (kk + 1 < 32)
            __builtin_prefetch(WoT + (size_t)tid * 8 + kk * 32, 0, 1);
        wait_async0();
        __syncthreads();
        v16h a = load_a(sA, 32, mt * 16, 0, lane);
#pragma unroll
        for (int j = 0; j < 4; ++j) {
            v16h bf = load_ct(sBT, 32, 0, (half * 4 + j) * 16, lane);
            acc[j].v = wmma32(a, bf, acc[j].v);
        }
        __syncthreads();
    }
#pragma unroll
    for (int j = 0; j < 4; ++j) {
        int col = (half * 4 + j) * 16 + ln;
#pragma unroll
        for (int r = 0; r < 8; ++r) {
            int row = mt * 16 + hi * 8 + r;
            out[(rbase + row) * F_ + col] = acc[j].f[r] + bo[col];
        }
    }
}

// ---------------------------------------------------------------------------
extern "C" void kernel_launch(void* const* d_in, const int* in_sizes, int n_in,
                              void* d_out, int out_size, void* d_ws, size_t ws_size,
                              hipStream_t stream) {
    const float* q_x     = (const float*)d_in[0];
    const float* kv_x    = (const float*)d_in[1];
    const float* t_embed = (const float*)d_in[2];
    const float* Wq  = (const float*)d_in[3];
    const float* Wk  = (const float*)d_in[4];
    const float* Wv  = (const float*)d_in[5];
    const float* Wqt = (const float*)d_in[6];
    const float* bqt = (const float*)d_in[7];
    const float* Wkt = (const float*)d_in[8];
    const float* bkt = (const float*)d_in[9];
    const float* Wvt = (const float*)d_in[10];
    const float* bvt = (const float*)d_in[11];
    const float* Wo  = (const float*)d_in[12];
    const float* bo  = (const float*)d_in[13];
    float* out = (float*)d_out;

    // workspace layout
    float* tbq = (float*)d_ws;                   // 262144 f32
    float* tbk = tbq + 262144;
    float* tbv = tbk + 262144;
    _Float16* attn = (_Float16*)(tbv + 262144);  //  4,194,304 f16
    _Float16* kvh  = attn + 4194304;             // 16,777,216 f16
    _Float16* qh   = kvh + 16777216;             //    524,288 f16
    _Float16* WqT  = qh + 524288;                //    131,072 f16 [h][n][k]
    _Float16* WkT  = WqT + 131072;
    _Float16* WvT  = WkT + 131072;
    _Float16* WoT  = WvT + 131072;               //    131,072 f16 [n][k]

    cvt16_kernel<<<16384, 256, 0, stream>>>(kv_x, kvh, 4194304);
    cvt16_kernel<<<512, 256, 0, stream>>>(q_x, qh, 131072);
    wtrans_kernel<<<1024, 128, 0, stream>>>(Wq, WqT);
    wtrans_kernel<<<1024, 128, 0, stream>>>(Wk, WkT);
    wtrans_kernel<<<1024, 128, 0, stream>>>(Wv, WvT);
    wotrans_kernel<<<128, 256, 0, stream>>>(Wo, WoT);
    bias_kernel<<<256, 128, 0, stream>>>(t_embed, Wqt, bqt, Wkt, bkt, Wvt, bvt,
                                         tbq, tbk, tbv);
    attn_kernel<<<512, 256, 0, stream>>>(qh, kvh, WqT, WkT, WvT,
                                         tbq, tbk, tbv, attn);
    outproj_kernel<<<64, 256, 0, stream>>>(attn, WoT, bo, out);
}
